// Qwen3SparseMoeBlock_12360915878735
// MI455X (gfx1250) — matrix operand order repaired and checked by reading, not compiled, not run on hardware
//
#include <hip/hip_runtime.h>
#include <hip/hip_bf16.h>

#define NUM_EXPERTS 64
#define TOP_K       8
#define HIDDEN      1024
#define INTER       768
#define T_TOKENS    1024
#define CAPACITY    256
#define KSTEP       64
#define ROW_PITCH   66   // 64 data dwords + 2 TDM pad dwords per row (bank-conflict free)

typedef __attribute__((ext_vector_type(16))) __bf16 v16bf;
typedef __attribute__((ext_vector_type(8)))  __bf16 v8bf;
typedef __attribute__((ext_vector_type(4)))  __bf16 v4bf;
typedef __attribute__((ext_vector_type(8)))  float  v8f;
typedef __attribute__((ext_vector_type(4)))  unsigned int v4u;
typedef __attribute__((ext_vector_type(8)))  int v8i;
typedef __attribute__((ext_vector_type(4)))  int v4i;

// ---------------------------------------------------------------------------
// Helpers
// ---------------------------------------------------------------------------
__device__ __forceinline__ v8bf cvt8_f32_to_bf16(const float* __restrict__ p) {
    float4 a = *(const float4*)(p);
    float4 b = *(const float4*)(p + 4);
    v8bf r;
    r[0] = (__bf16)a.x; r[1] = (__bf16)a.y; r[2] = (__bf16)a.z; r[3] = (__bf16)a.w;
    r[4] = (__bf16)b.x; r[5] = (__bf16)b.y; r[6] = (__bf16)b.z; r[7] = (__bf16)b.w;
    return r;
}

union FragBF16 { v16bf v; v8bf h[2]; };

// LDS dword index of float f within a TDM-padded row (1 pad per 32 data dwords)
__device__ __forceinline__ int lds_idx(int f) { return f + (f >> 5); }

// Tensor Data Mover: DMA a [tile1 rows x KSTEP floats] fp32 tile into LDS,
// inserting 1 pad dword after every 32 data dwords (row pitch = 66 dwords).
// D# layout per CDNA5 ISA. 6-arg builtin variant.
__device__ __forceinline__ void tdm_load_tile(
    unsigned int lds_off, const float* gptr,
    unsigned int tdim0, unsigned int tdim1, unsigned int stride0,
    unsigned int tile0, unsigned int tile1)
{
    unsigned long long ga = (unsigned long long)(const void*)gptr;
    v4u g0;
    g0[0] = 1u;                                   // count=1, user mode, no gather
    g0[1] = lds_off;                              // LDS byte address
    g0[2] = (unsigned int)ga;                     // global_addr[31:0]
    g0[3] = (unsigned int)((ga >> 32) & 0x1FFFFFFu) | (2u << 30);  // addr[56:32] | type=2
    v8i g1;
    g1[0] = (int)((2u << 16)                      // data_size = 4B
                | (1u << 20)                      // pad_enable
                | (4u << 22));                    // pad_interval: code 4 = 32 dwords
                                                  // pad_amount:   code 0 = 1 dword
    g1[1] = (int)((tdim0 & 0xFFFFu) << 16);       // tensor_dim0[15:0] @ bit48
    g1[2] = (int)(((tdim0 >> 16) & 0xFFFFu) | ((tdim1 & 0xFFFFu) << 16));
    g1[3] = (int)(((tdim1 >> 16) & 0xFFFFu) | ((tile0 & 0xFFFFu) << 16));
    g1[4] = (int)(tile1 & 0xFFFFu);               // tile_dim1, tile_dim2=0
    g1[5] = (int)stride0;                         // tensor_dim0_stride[31:0]
    g1[6] = 0;                                    // stride0[47:32]=0, dim1_stride lo=0
    g1[7] = 0;
    v4i z4 = {0, 0, 0, 0};
    v8i z8 = {0, 0, 0, 0, 0, 0, 0, 0};
    __builtin_amdgcn_tensor_load_to_lds(g0, g1, z4, z4, z8, 0);
}

__device__ __forceinline__ unsigned int lds_offset_of(const void* p) {
    return (unsigned int)(unsigned long long)p;    // low 32 bits = LDS byte offset
}

// ---------------------------------------------------------------------------
// 1) Router: logits -> softmax -> top-8 -> renormalize
// ---------------------------------------------------------------------------
__global__ __launch_bounds__(64) void router_kernel(
    const float* __restrict__ x, const float* __restrict__ gate_w,
    int* __restrict__ topk_idx, float* __restrict__ topk_w)
{
    const int t = blockIdx.x;
    const int e = threadIdx.x;
    __shared__ float probs[NUM_EXPERTS];
    __shared__ float red;

    const float* xr = x + (size_t)t * HIDDEN;
    const float* wr = gate_w + (size_t)e * HIDDEN;
    float acc = 0.f;
    #pragma unroll 4
    for (int h = 0; h < HIDDEN; h += 4) {
        float4 xv = *(const float4*)(xr + h);
        float4 wv = *(const float4*)(wr + h);
        acc += xv.x * wv.x + xv.y * wv.y + xv.z * wv.z + xv.w * wv.w;
    }
    probs[e] = acc;
    __syncthreads();
    if (e == 0) {
        float mx = probs[0];
        for (int j = 1; j < NUM_EXPERTS; ++j) mx = probs[j] > mx ? probs[j] : mx;
        red = mx;
    }
    __syncthreads();
    float p = __expf(acc - red);
    probs[e] = p;
    __syncthreads();
    if (e == 0) {
        float s = 0.f;
        for (int j = 0; j < NUM_EXPERTS; ++j) s += probs[j];
        red = s;
    }
    __syncthreads();
    probs[e] = p * __builtin_amdgcn_rcpf(red);
    __syncthreads();
    if (e == 0) {
        int   idxs[TOP_K];
        float vals[TOP_K];
        float sum8 = 0.f;
        for (int k = 0; k < TOP_K; ++k) {
            float best = -1.f; int bi = 0;
            for (int j = 0; j < NUM_EXPERTS; ++j)
                if (probs[j] > best) { best = probs[j]; bi = j; }
            idxs[k] = bi; vals[k] = best; sum8 += best;
            probs[bi] = -1.f;
        }
        const float inv = __builtin_amdgcn_rcpf(sum8);
        for (int k = 0; k < TOP_K; ++k) {
            topk_idx[t * TOP_K + k] = idxs[k];
            topk_w[t * TOP_K + k]   = vals[k] * inv;
        }
    }
}

// ---------------------------------------------------------------------------
// 2) Convert activations fp32 -> bf16
// ---------------------------------------------------------------------------
__global__ __launch_bounds__(256) void cvt_x_kernel(
    const float* __restrict__ x, __bf16* __restrict__ xb)
{
    const int i = (blockIdx.x * 256 + threadIdx.x) * 4;
    float4 v = *(const float4*)(x + i);
    v4bf o;
    o[0] = (__bf16)v.x; o[1] = (__bf16)v.y; o[2] = (__bf16)v.z; o[3] = (__bf16)v.w;
    *(v4bf*)(xb + i) = o;
}

// ---------------------------------------------------------------------------
// 3) Zero expert counters
// ---------------------------------------------------------------------------
__global__ void zero_counts_kernel(int* __restrict__ counts)
{
    if (threadIdx.x < NUM_EXPERTS) counts[threadIdx.x] = 0;
}

// ---------------------------------------------------------------------------
// 4) Dispatch: (token,k) -> (expert, slot)
// ---------------------------------------------------------------------------
__global__ __launch_bounds__(256) void dispatch_kernel(
    const int* __restrict__ topk_idx, int* __restrict__ counts,
    int* __restrict__ slot_assign, int* __restrict__ assign_pos)
{
    const int a = blockIdx.x * 256 + threadIdx.x;
    const int e = topk_idx[a];
    const int p = atomicAdd(&counts[e], 1);
    assign_pos[a] = p;
    if (p < CAPACITY) slot_assign[e * CAPACITY + p] = a;
}

// ---------------------------------------------------------------------------
// 5) Fused gate+up GEMM, TDM-staged weight tiles, silu(g)*u epilogue.
//    Block: 256 threads = 8 waves (2 M x 4 N); block tile 32 x 64.
//    Weight tiles [64 x 64] fp32 double-buffered in LDS via tensor_load_to_lds.
//    4 WMMAs per wave per barrier.
// ---------------------------------------------------------------------------
__global__ __launch_bounds__(256) void gateup_kernel(
    const __bf16* __restrict__ xb, const float* __restrict__ wg,
    const float* __restrict__ wu, const int* __restrict__ counts,
    const int* __restrict__ slot_assign, __bf16* __restrict__ hact)
{
    __shared__ __align__(16) float ldsG[2][64 * ROW_PITCH];
    __shared__ __align__(16) float ldsU[2][64 * ROW_PITCH];

    const int e = blockIdx.z;
    int cnt = counts[e];
    if (cnt > CAPACITY) cnt = CAPACITY;
    const int mTile = blockIdx.y * 32;
    if (mTile >= cnt) return;                      // block-uniform early exit

    const int lane = threadIdx.x & 31;
    const int wid  = threadIdx.x >> 5;
    const int sel  = lane >> 4;
    const int l15  = lane & 15;
    const int mBase = mTile + (wid >> 2) * 16;
    const int nTile = blockIdx.x * 64;
    const int nLoc  = (wid & 3) * 16 + l15;        // 0..63 within block N tile

    // A gather: dispatched token row (bf16)
    const int row = mBase + l15;
    const int aid = (row < cnt) ? slot_assign[e * CAPACITY + row] : 0;
    const int tok = aid >> 3;
    const __bf16* xrow = xb + (size_t)tok * HIDDEN;

    const float* wgTile = wg + ((size_t)e * INTER + nTile) * HIDDEN;
    const float* wuTile = wu + ((size_t)e * INTER + nTile) * HIDDEN;

    // Preload k=0 tiles via TDM (wave 0), then barrier.
    if (wid == 0) {
        tdm_load_tile(lds_offset_of(&ldsG[0][0]), wgTile, HIDDEN, INTER, HIDDEN, KSTEP, 64);
        tdm_load_tile(lds_offset_of(&ldsU[0][0]), wuTile, HIDDEN, INTER, HIDDEN, KSTEP, 64);
        __builtin_amdgcn_s_wait_tensorcnt(0);
    }
    __syncthreads();

    v8f accG = {}; v8f accU = {};
    int cur = 0;
    for (int k = 0; k < HIDDEN; k += KSTEP) {
        const int nxt = cur ^ 1;
        if (wid == 0 && (k + KSTEP) < HIDDEN) {    // prefetch next K tile
            tdm_load_tile(lds_offset_of(&ldsG[nxt][0]), wgTile + (k + KSTEP),
                          HIDDEN, INTER, HIDDEN, KSTEP, 64);
            tdm_load_tile(lds_offset_of(&ldsU[nxt][0]), wuTile + (k + KSTEP),
                          HIDDEN, INTER, HIDDEN, KSTEP, 64);
        }

        #pragma unroll
        for (int kk = 0; kk < KSTEP; kk += 32) {
            const int kA0 = k + kk + sel * 16;
            const int kA1 = k + kk + sel * 16 + 8;
            FragBF16 A, Bg, Bu;
            A.h[0] = *(const v8bf*)(xrow + kA0);
            A.h[1] = *(const v8bf*)(xrow + kA1);
            const int bOff = nLoc * ROW_PITCH + lds_idx(kk + sel * 16);
            const float* bg = &ldsG[cur][bOff];
            const float* bu = &ldsU[cur][bOff];
            Bg.h[0] = cvt8_f32_to_bf16(bg); Bg.h[1] = cvt8_f32_to_bf16(bg + 8);
            Bu.h[0] = cvt8_f32_to_bf16(bu); Bu.h[1] = cvt8_f32_to_bf16(bu + 8);
            accG = __builtin_amdgcn_wmma_f32_16x16x32_bf16(
                       false, A.v, false, Bg.v, (short)0, accG, false, false);
            accU = __builtin_amdgcn_wmma_f32_16x16x32_bf16(
                       false, A.v, false, Bu.v, (short)0, accU, false, false);
        }

        if (wid == 0) __builtin_amdgcn_s_wait_tensorcnt(0);
        __syncthreads();
        cur = nxt;
    }

    // Epilogue: silu(g) * u -> bf16  (C/D layout: M = r + 8*sel, N = l15)
    const int outRow = mBase + 8 * sel;
    const int col    = nTile + nLoc;
    #pragma unroll
    for (int r = 0; r < 8; ++r) {
        const float g = accG[r];
        const float u = accU[r];
        const float s = g * __builtin_amdgcn_rcpf(1.f + __expf(-g));
        hact[((size_t)e * CAPACITY + (outRow + r)) * INTER + col] = (__bf16)(s * u);
    }
}

// ---------------------------------------------------------------------------
// 6) Down GEMM with TDM-staged w_down tiles: y[e,c,:H] = hact @ w_down[e]^T
// ---------------------------------------------------------------------------
__global__ __launch_bounds__(256) void down_kernel(
    const __bf16* __restrict__ hact, const float* __restrict__ wd,
    const int* __restrict__ counts, float* __restrict__ y)
{
    __shared__ __align__(16) float ldsB[2][64 * ROW_PITCH];

    const int e = blockIdx.z;
    int cnt = counts[e];
    if (cnt > CAPACITY) cnt = CAPACITY;
    const int mTile = blockIdx.y * 32;
    if (mTile >= cnt) return;

    const int lane = threadIdx.x & 31;
    const int wid  = threadIdx.x >> 5;
    const int sel  = lane >> 4;
    const int l15  = lane & 15;
    const int mBase = mTile + (wid >> 2) * 16;
    const int nTile = blockIdx.x * 64;
    const int nLoc  = (wid & 3) * 16 + l15;

    const __bf16* arow   = hact + ((size_t)e * CAPACITY + mBase + l15) * INTER;
    const float*  wdTile = wd + ((size_t)e * HIDDEN + nTile) * INTER;

    if (wid == 0) {
        tdm_load_tile(lds_offset_of(&ldsB[0][0]), wdTile, INTER, HIDDEN, INTER, KSTEP, 64);
        __builtin_amdgcn_s_wait_tensorcnt(0);
    }
    __syncthreads();

    v8f acc = {};
    int cur = 0;
    for (int k = 0; k < INTER; k += KSTEP) {
        const int nxt = cur ^ 1;
        if (wid == 0 && (k + KSTEP) < INTER) {
            tdm_load_tile(lds_offset_of(&ldsB[nxt][0]), wdTile + (k + KSTEP),
                          INTER, HIDDEN, INTER, KSTEP, 64);
        }

        #pragma unroll
        for (int kk = 0; kk < KSTEP; kk += 32) {
            const int kA0 = k + kk + sel * 16;
            const int kA1 = k + kk + sel * 16 + 8;
            FragBF16 A, B;
            A.h[0] = *(const v8bf*)(arow + kA0);
            A.h[1] = *(const v8bf*)(arow + kA1);
            const float* bp = &ldsB[cur][nLoc * ROW_PITCH + lds_idx(kk + sel * 16)];
            B.h[0] = cvt8_f32_to_bf16(bp);
            B.h[1] = cvt8_f32_to_bf16(bp + 8);
            acc = __builtin_amdgcn_wmma_f32_16x16x32_bf16(
                      false, A.v, false, B.v, (short)0, acc, false, false);
        }

        if (wid == 0) __builtin_amdgcn_s_wait_tensorcnt(0);
        __syncthreads();
        cur = nxt;
    }

    const int outRow = mBase + 8 * sel;
    const int col    = nTile + nLoc;
    #pragma unroll
    for (int r = 0; r < 8; ++r)
        y[((size_t)e * CAPACITY + (outRow + r)) * HIDDEN + col] = acc[r];
}

// ---------------------------------------------------------------------------
// 7) Combine: deterministic weighted gather-sum back to tokens
// ---------------------------------------------------------------------------
__global__ __launch_bounds__(256) void combine_kernel(
    const int* __restrict__ topk_idx, const float* __restrict__ topk_w,
    const int* __restrict__ assign_pos, const float* __restrict__ y,
    float* __restrict__ out)
{
    const int t = blockIdx.x;
    const int h = threadIdx.x * 4;
    float4 acc = {0.f, 0.f, 0.f, 0.f};
    #pragma unroll
    for (int k = 0; k < TOP_K; ++k) {
        const int a = t * TOP_K + k;
        const int p = assign_pos[a];
        if (p < CAPACITY) {
            const int   e = topk_idx[a];
            const float w = topk_w[a];
            float4 v = *(const float4*)(y + ((size_t)e * CAPACITY + p) * HIDDEN + h);
            acc.x += w * v.x; acc.y += w * v.y; acc.z += w * v.z; acc.w += w * v.w;
        }
    }
    *(float4*)(out + (size_t)t * HIDDEN + h) = acc;
}

// ---------------------------------------------------------------------------
// Launch
// ---------------------------------------------------------------------------
extern "C" void kernel_launch(void* const* d_in, const int* in_sizes, int n_in,
                              void* d_out, int out_size, void* d_ws, size_t ws_size,
                              hipStream_t stream)
{
    const float* x      = (const float*)d_in[0];  // [1,1024,1024]
    const float* gate_w = (const float*)d_in[1];  // [64,1024]
    const float* w_gate = (const float*)d_in[2];  // [64,768,1024]
    const float* w_up   = (const float*)d_in[3];  // [64,768,1024]
    const float* w_down = (const float*)d_in[4];  // [64,1024,768]
    float* out = (float*)d_out;

    char* ws = (char*)d_ws;
    auto alloc = [&](size_t bytes) -> char* {
        char* p = ws;
        ws += (bytes + 255) & ~(size_t)255;
        return p;
    };
    __bf16* xb          = (__bf16*)alloc((size_t)T_TOKENS * HIDDEN * 2);
    int*    counts      = (int*)   alloc(NUM_EXPERTS * 4);
    int*    slot_assign = (int*)   alloc((size_t)NUM_EXPERTS * CAPACITY * 4);
    int*    assign_pos  = (int*)   alloc((size_t)T_TOKENS * TOP_K * 4);
    int*    topk_idx    = (int*)   alloc((size_t)T_TOKENS * TOP_K * 4);
    float*  topk_w      = (float*) alloc((size_t)T_TOKENS * TOP_K * 4);
    __bf16* hact        = (__bf16*)alloc((size_t)NUM_EXPERTS * CAPACITY * INTER * 2);
    float*  y           = (float*) alloc((size_t)NUM_EXPERTS * CAPACITY * HIDDEN * 4);

    router_kernel<<<dim3(T_TOKENS), dim3(64), 0, stream>>>(x, gate_w, topk_idx, topk_w);
    cvt_x_kernel<<<dim3((T_TOKENS * HIDDEN) / 1024), dim3(256), 0, stream>>>(x, xb);
    zero_counts_kernel<<<dim3(1), dim3(64), 0, stream>>>(counts);
    dispatch_kernel<<<dim3((T_TOKENS * TOP_K) / 256), dim3(256), 0, stream>>>(
        topk_idx, counts, slot_assign, assign_pos);
    gateup_kernel<<<dim3(INTER / 64, CAPACITY / 32, NUM_EXPERTS), dim3(256), 0, stream>>>(
        xb, w_gate, w_up, counts, slot_assign, hact);
    down_kernel<<<dim3(HIDDEN / 64, CAPACITY / 32, NUM_EXPERTS), dim3(256), 0, stream>>>(
        hact, w_down, counts, y);
    combine_kernel<<<dim3(T_TOKENS), dim3(256), 0, stream>>>(
        topk_idx, topk_w, assign_pos, y, out);
}
